// IobesNERDecoder_63307817943771
// MI455X (gfx1250) — matrix operand order repaired and checked by reading, not compiled, not run on hardware
//
#include <hip/hip_runtime.h>

// ---------------------------------------------------------------------------
// IOBES NER AR-GRU decoder for MI455X (gfx1250, wave32, WMMA).
//
// Decomposition:
//   precompute:  gx_w[b*T+t, 3H] = w[b,t,:] @ Wih_word^T            (big WMMA GEMM)
//                labelGx[l, 3H]  = label_emb[l] @ Wih_label^T + b_ih + b_hh
//   sequential:  per step t (persistent kernel, grid barrier):
//     A) gh = h_bf16 @ Whh^T (WMMA bf16, f32 acc); GRU gating -> h_new
//     B) logits = h_new @ Wout^T + b_out + mask[prev]; argmax -> pred (feedback)
// ---------------------------------------------------------------------------

typedef unsigned short u16_t;
typedef __attribute__((ext_vector_type(16))) __bf16        bf16x16;
typedef __attribute__((ext_vector_type(8)))  float         f32x8;
typedef __attribute__((ext_vector_type(4)))  unsigned int  u32x4;

union BFrag { u32x4 u[2]; bf16x16 v; };   // 16 bf16 = two 16B global loads

#define CB   128        // batch
#define CT   512        // seq len
#define CH   1024       // hidden
#define CE   128        // label emb
#define CL   17         // labels
#define CH3  3072       // 3*H
#define CEH  1152       // E+H
#define NBLK 64         // persistent grid (co-resident on MI455X)

__device__ __forceinline__ u16_t f2bf(float f) {           // RNE f32 -> bf16
    unsigned int u = __builtin_bit_cast(unsigned int, f);
    unsigned int r = u + 0x7FFFu + ((u >> 16) & 1u);
    return (u16_t)(r >> 16);
}
__device__ __forceinline__ float fsigmoid(float x) { return 1.f / (1.f + __expf(-x)); }
__device__ __forceinline__ float ftanh(float x)    { return 2.f / (1.f + __expf(-2.f * x)) - 1.f; }

__device__ __forceinline__ f32x8 wmma_bf16(bf16x16 a, bf16x16 b, f32x8 c) {
    return __builtin_amdgcn_wmma_f32_16x16x32_bf16(false, a, false, b, (short)0, c, false, false);
}

// LABELS = ["O"] + [p-t for t in 0..3 for p in B,I,E,S]; id = 1 + 4*type + p
__device__ __forceinline__ float mask_val(int prev, int nxt) {
    const float NEG = -1e12f;
    if (prev == 0 || (((prev - 1) & 3) >= 2)) {            // after O/E/S: allow O,B,S
        if (nxt == 0) return 0.f;
        int p = (nxt - 1) & 3;
        return (p == 0 || p == 3) ? 0.f : NEG;
    } else {                                               // after B/I-tag: only I/E same tag
        if (nxt == 0) return NEG;
        int tag = (prev - 1) >> 2;
        int nt = (nxt - 1) >> 2, np = (nxt - 1) & 3;
        return (nt == tag && (np == 1 || np == 2)) ? 0.f : NEG;
    }
}

// Two-phase generation-count grid barrier (device scope atomics + s_sleep spin).
__device__ __forceinline__ void grid_barrier(unsigned* bar, unsigned nblk) {
    __syncthreads();
    if (threadIdx.x == 0) {
        __threadfence();
        unsigned gen  = __hip_atomic_load(&bar[1], __ATOMIC_ACQUIRE, __HIP_MEMORY_SCOPE_AGENT);
        unsigned prev = __hip_atomic_fetch_add(&bar[0], 1u, __ATOMIC_ACQ_REL, __HIP_MEMORY_SCOPE_AGENT);
        if (prev == nblk - 1) {
            __hip_atomic_store(&bar[0], 0u, __ATOMIC_RELAXED, __HIP_MEMORY_SCOPE_AGENT);
            __hip_atomic_store(&bar[1], gen + 1u, __ATOMIC_RELEASE, __HIP_MEMORY_SCOPE_AGENT);
        } else {
            while (__hip_atomic_load(&bar[1], __ATOMIC_ACQUIRE, __HIP_MEMORY_SCOPE_AGENT) == gen)
                __builtin_amdgcn_s_sleep(8);
        }
        __threadfence();
    }
    __syncthreads();
}

// ---------------------------------------------------------------------------
__global__ void init_kernel(unsigned* bar, int* pred, float* hf32, u16_t* hbf) {
    int i = blockIdx.x * blockDim.x + threadIdx.x;
    int s = gridDim.x * blockDim.x;
    for (int k = i; k < 64; k += s) bar[k] = 0u;
    for (int k = i; k < CB; k += s) pred[k] = 0;           // start label 'O'
    for (int k = i; k < 2 * CB * CH; k += s) { hf32[k] = 0.f; hbf[k] = 0; }
}

__global__ void cvt_bf16_kernel(u16_t* dst, const float* src, long long n) {
    long long i = (long long)blockIdx.x * blockDim.x + threadIdx.x;
    long long s = (long long)gridDim.x * blockDim.x;
    for (; i < n; i += s) dst[i] = f2bf(src[i]);
}

__global__ void cvt_wihw_kernel(u16_t* dst, const float* Wih) {   // word part of W_ih
    long long i = (long long)blockIdx.x * blockDim.x + threadIdx.x;
    long long s = (long long)gridDim.x * blockDim.x;
    for (; i < (long long)CH3 * CH; i += s) {
        long long j = i >> 10, k = i & 1023;
        dst[i] = f2bf(Wih[j * CEH + CE + k]);
    }
}

__global__ void labelgx_kernel(float* out, const float* lemb, const float* Wih,
                               const float* bih, const float* bhh) {
    int i = blockIdx.x * blockDim.x + threadIdx.x;
    if (i >= CL * CH3) return;
    int l = i / CH3, j = i - l * CH3;
    float s = bih[j] + bhh[j];                              // fold both biases
    const float* wr = Wih + (long long)j * CEH;
    const float* le = lemb + l * CE;
    #pragma unroll 8
    for (int e = 0; e < CE; ++e) s += le[e] * wr[e];
    out[i] = s;
}

// ---------------------------------------------------------------------------
// gx_w[M=B*T, N=3H] = w_bf16[M, K=H] x Wihw[N, K]^T  (412 GFLOP, WMMA bf16)
// Tile 128x128 per WG; 8 waves, each wave 64x32 (4x2 accumulators).
__global__ void __launch_bounds__(256) gemm_gxw_kernel(float* __restrict__ gxw,
                                                       const u16_t* __restrict__ A,
                                                       const u16_t* __restrict__ Bw) {
    const int wgid = blockIdx.x;
    const int nt = wgid % (CH3 / 128), mt = wgid / (CH3 / 128);
    const int lane = threadIdx.x & 31, wave = threadIdx.x >> 5;
    const int mhalf = wave >> 2, nq = wave & 3;
    const int m0 = mt * 128 + mhalf * 64;
    const int c0 = nt * 128 + nq * 32;
    const int koffA = (lane >> 4) * 16;                    // A: lanes 16-31 hold K+16 half
    const int koffB = (lane >> 4) * 16;                    // B: lanes 16-31 hold K+16 half
    const u16_t* ap[4]; const u16_t* bp[2];
    #pragma unroll
    for (int a = 0; a < 4; ++a)
        ap[a] = A + (long long)(m0 + a * 16 + (lane & 15)) * CH + koffA;
    #pragma unroll
    for (int b = 0; b < 2; ++b)
        bp[b] = Bw + (long long)(c0 + b * 16 + (lane & 15)) * CH + koffB;

    const f32x8 z8 = {0.f, 0.f, 0.f, 0.f, 0.f, 0.f, 0.f, 0.f};
    f32x8 acc[4][2];
    #pragma unroll
    for (int a = 0; a < 4; ++a)
        #pragma unroll
        for (int b = 0; b < 2; ++b) acc[a][b] = z8;

    for (int k = 0; k < CH; k += 32) {
        __builtin_prefetch(ap[0] + k + 256, 0, 1);         // global_prefetch_b8
        BFrag af[4], bf[2];
        #pragma unroll
        for (int a = 0; a < 4; ++a) {
            af[a].u[0] = *(const u32x4*)(ap[a] + k);
            af[a].u[1] = *(const u32x4*)(ap[a] + k + 8);
        }
        #pragma unroll
        for (int b = 0; b < 2; ++b) {
            bf[b].u[0] = *(const u32x4*)(bp[b] + k);
            bf[b].u[1] = *(const u32x4*)(bp[b] + k + 8);
        }
        #pragma unroll
        for (int a = 0; a < 4; ++a)
            #pragma unroll
            for (int b = 0; b < 2; ++b)
                acc[a][b] = wmma_bf16(af[a].v, bf[b].v, acc[a][b]);
    }
    const int mofs = (lane >> 4) * 8;                      // C: VGPR i -> M = i or i+8
    #pragma unroll
    for (int a = 0; a < 4; ++a)
        #pragma unroll
        for (int b = 0; b < 2; ++b)
            #pragma unroll
            for (int i = 0; i < 8; ++i)
                gxw[(long long)(m0 + a * 16 + i + mofs) * CH3 + (c0 + b * 16 + (lane & 15))]
                    = acc[a][b][i];
}

// ---------------------------------------------------------------------------
// Persistent AR decode: 64 WGs x 256 threads, 512 steps, 2 grid barriers/step.
__global__ void __launch_bounds__(256) decode_kernel(
    const float* __restrict__ gxw, const float* __restrict__ labelGx,
    const u16_t* __restrict__ WhhB, const float* __restrict__ Wout,
    const float* __restrict__ bout,
    float* __restrict__ hf32, u16_t* __restrict__ hbf, int* __restrict__ pred,
    unsigned* __restrict__ bar, float* __restrict__ out) {

    const int wg = blockIdx.x, lane = threadIdx.x & 31, wave = threadIdx.x >> 5;
    const int btile = wg & 1, jt = wg >> 1;                // 2 b-tiles x 32 j-tiles
    const int msub = wave >> 1, nsub = wave & 1;           // 8 waves: 4x2 over 64x32 tile
    const int row0 = btile * 64 + msub * 16;               // batch rows of this wave
    const int col0 = jt * 32 + nsub * 16;                  // hidden cols of this wave
    const int nlane = lane & 15;
    const int koffA = (lane >> 4) * 16;
    const int koffB = (lane >> 4) * 16;
    const long long aBase = (long long)(row0 + nlane) * CH + koffA;
    const u16_t* Bp[3];                                    // r,z,n gate weight rows
    #pragma unroll
    for (int g = 0; g < 3; ++g)
        Bp[g] = WhhB + (long long)(g * CH + col0 + nlane) * CH + koffB;
    const f32x8 z8 = {0.f, 0.f, 0.f, 0.f, 0.f, 0.f, 0.f, 0.f};

    for (int t = 0; t < CT; ++t) {
        // ---- phase A: gh = h @ Whh^T (WMMA), gating, h_new -----------------
        const u16_t* hb = hbf + (size_t)(t & 1) * (CB * CH);
        f32x8 accR = z8, accZ = z8, accN = z8;
        for (int k = 0; k < CH; k += 32) {
            __builtin_prefetch(Bp[0] + k + 128, 0, 1);
            BFrag A, Br, Bz, Bn;
            A.u[0]  = *(const u32x4*)(hb + aBase + k);
            A.u[1]  = *(const u32x4*)(hb + aBase + k + 8);
            Br.u[0] = *(const u32x4*)(Bp[0] + k);  Br.u[1] = *(const u32x4*)(Bp[0] + k + 8);
            Bz.u[0] = *(const u32x4*)(Bp[1] + k);  Bz.u[1] = *(const u32x4*)(Bp[1] + k + 8);
            Bn.u[0] = *(const u32x4*)(Bp[2] + k);  Bn.u[1] = *(const u32x4*)(Bp[2] + k + 8);
            accR = wmma_bf16(A.v, Br.v, accR);
            accZ = wmma_bf16(A.v, Bz.v, accZ);
            accN = wmma_bf16(A.v, Bn.v, accN);
        }
        const float* hprev  = hf32 + (size_t)(t & 1) * (CB * CH);
        float*       hnew   = hf32 + (size_t)((t + 1) & 1) * (CB * CH);
        u16_t*       hbnew  = hbf  + (size_t)((t + 1) & 1) * (CB * CH);
        const int jcol = col0 + nlane;
        const int mofs = (lane >> 4) * 8;
        #pragma unroll
        for (int i = 0; i < 8; ++i) {
            const int b = row0 + i + mofs;
            const long long gbase = ((long long)b * CT + t) * CH3;
            const float* lg = labelGx + (long long)pred[b] * CH3;
            float gr = gxw[gbase + jcol]            + lg[jcol];
            float gz = gxw[gbase + CH + jcol]       + lg[CH + jcol];
            float gn = gxw[gbase + 2 * CH + jcol]   + lg[2 * CH + jcol];
            float r  = fsigmoid(gr + accR[i]);
            float zz = fsigmoid(gz + accZ[i]);
            float nn = ftanh(gn + r * accN[i]);
            float hv = (1.f - zz) * nn + zz * hprev[b * CH + jcol];
            hnew[b * CH + jcol]  = hv;
            hbnew[b * CH + jcol] = f2bf(hv);
        }
        grid_barrier(bar, NBLK);

        // ---- phase B: logits + mask + argmax, one wave per batch row -------
        if (wg < 16) {
            const int b = wg * 8 + wave;
            float acc[CL];
            #pragma unroll
            for (int l = 0; l < CL; ++l) acc[l] = 0.f;
            const float* hrow = hnew + b * CH;
            for (int jj = 0; jj < CH / 32; ++jj) {
                int j = jj * 32 + lane;
                float hv = hrow[j];
                #pragma unroll
                for (int l = 0; l < CL; ++l) acc[l] += hv * Wout[l * CH + j];
            }
            #pragma unroll
            for (int l = 0; l < CL; ++l) {
                acc[l] += __shfl_xor(acc[l], 16);
                acc[l] += __shfl_xor(acc[l], 8);
                acc[l] += __shfl_xor(acc[l], 4);
                acc[l] += __shfl_xor(acc[l], 2);
                acc[l] += __shfl_xor(acc[l], 1);
            }
            if (lane == 0) {
                const int pv = pred[b];
                float best = -3.4e38f; int bi = 0;
                float* orow = out + ((long long)b * CT + t) * CL;
                #pragma unroll
                for (int l = 0; l < CL; ++l) {
                    float v = acc[l] + bout[l] + mask_val(pv, l);
                    orow[l] = v;
                    if (v > best) { best = v; bi = l; }    // strict > == first argmax
                }
                pred[b] = bi;
            }
        }
        grid_barrier(bar, NBLK);
    }
}

// ---------------------------------------------------------------------------
extern "C" void kernel_launch(void* const* d_in, const int* in_sizes, int n_in,
                              void* d_out, int out_size, void* d_ws, size_t ws_size,
                              hipStream_t stream) {
    const float* wemb = (const float*)d_in[0];   // [B,T,H]
    const float* lemb = (const float*)d_in[1];   // [17,E]
    const float* Wih  = (const float*)d_in[2];   // [3H, E+H]
    const float* Whh  = (const float*)d_in[3];   // [3H, H]
    const float* bih  = (const float*)d_in[4];
    const float* bhh  = (const float*)d_in[5];
    const float* Wout = (const float*)d_in[6];   // [17, H]
    const float* bout = (const float*)d_in[7];
    float* out = (float*)d_out;                  // [B,T,17]

    char* ws = (char*)d_ws;
    size_t off = 0;
    auto take = [&](size_t bytes) -> char* {
        char* p = ws + off;
        off = (off + bytes + 255) & ~(size_t)255;
        return p;
    };
    const long long BT = (long long)CB * CT;
    unsigned* bar    = (unsigned*)take(256);
    int*      pred   = (int*)     take(CB * sizeof(int));
    float*    hf32   = (float*)   take((size_t)2 * CB * CH * sizeof(float));
    u16_t*    hbf    = (u16_t*)   take((size_t)2 * CB * CH * sizeof(u16_t));
    float*    lgx    = (float*)   take((size_t)CL * CH3 * sizeof(float));
    u16_t*    whhB   = (u16_t*)   take((size_t)CH3 * CH * sizeof(u16_t));
    u16_t*    wihwB  = (u16_t*)   take((size_t)CH3 * CH * sizeof(u16_t));
    u16_t*    wB     = (u16_t*)   take((size_t)BT * CH * sizeof(u16_t));
    float*    gxw    = (float*)   take((size_t)BT * CH3 * sizeof(float));
    (void)ws_size; (void)in_sizes; (void)n_in; (void)out_size;

    init_kernel<<<512, 256, 0, stream>>>(bar, pred, hf32, hbf);
    cvt_bf16_kernel<<<2048, 256, 0, stream>>>(whhB, Whh, (long long)CH3 * CH);
    cvt_wihw_kernel<<<2048, 256, 0, stream>>>(wihwB, Wih);
    cvt_bf16_kernel<<<4096, 256, 0, stream>>>(wB, wemb, BT * CH);
    labelgx_kernel<<<(CL * CH3 + 255) / 256, 256, 0, stream>>>(lgx, lemb, Wih, bih, bhh);
    gemm_gxw_kernel<<<(int)((BT / 128) * (CH3 / 128)), 256, 0, stream>>>(gxw, wB, wihwB);
    decode_kernel<<<NBLK, 256, 0, stream>>>(gxw, lgx, whhB, Wout, bout,
                                            hf32, hbf, pred, bar, out);
}
